// sampeling_pattern_intersection_28913719837066
// MI455X (gfx1250) — matrix operand order repaired and checked by reading, not compiled, not run on hardware
//
#include <hip/hip_runtime.h>
#include <cstdint>
#include <cstddef>

// ---------------------------------------------------------------------------
// MI455X (gfx1250) U-Net: implicit-GEMM conv3x3 on v_wmma_f32_16x16x32_f16.
// - Activations NHWC f16 (C padded to multiples of 32): A-fragments are two
//   contiguous 16B global loads per lane.
// - Weights pre-packed to B-fragment layout; per K-chunk the block stages
//   them into LDS with global_load_async_to_lds_b128 (ASYNCcnt) and all 8
//   waves consume via ds_load — 8x less B global traffic.
// - Each wave computes a 16Mx32N tile (2 accumulators) to halve A traffic.
// - Norm+ReLU fused into one pass writing straight into concat buffers.
// ---------------------------------------------------------------------------

typedef _Float16 half_t;
typedef __attribute__((ext_vector_type(16))) _Float16 v16h;
typedef __attribute__((ext_vector_type(8)))  _Float16 v8h;
typedef __attribute__((ext_vector_type(8)))  float    v8f;

// ------------------------- input conversion --------------------------------
__global__ void k_convert_input(const float* __restrict__ ks,
                                half_t* __restrict__ dst, int npix) {
  int t = blockIdx.x * blockDim.x + threadIdx.x;
  if (t >= npix * 32) return;
  int pix = t >> 5, ch = t & 31;
  float v = (ch < 2) ? ks[(size_t)pix * 2 + ch] : 0.f;
  dst[(size_t)pix * 32 + ch] = (half_t)v;
}

// ------------------------- weight pre-pack ---------------------------------
// OIHW f32 -> f16 B-fragments, chunk-major so one (nt,chunk) stage is a
// contiguous 9*512-half block:
//   e = (((nt*chunks + c)*9 + t)*32 + lane)*16 + i
// B-matrix layout (32x16 f16): lane -> N = lane&15, K = (lane>>4)*8 + (i&7) + (i>>3)*16.
__global__ void k_prep_w(const float* __restrict__ w, half_t* __restrict__ out,
                         int Cout, int Cin, int chunks, int total) {
  int e = blockIdx.x * blockDim.x + threadIdx.x;
  if (e >= total) return;
  int i    = e & 15;
  int lane = (e >> 4) & 31;
  int rest = e >> 9;
  int t  = rest % 9; rest /= 9;
  int c  = rest % chunks;
  int nt = rest / chunks;
  int n   = nt * 16 + (lane & 15);
  int cin = c * 32 + ((lane >> 4) << 3) + (i & 7) + ((i >> 3) << 4);
  float v = 0.f;
  if (cin < Cin && n < Cout) v = w[((size_t)n * Cin + cin) * 9 + t];
  out[e] = (half_t)v;
}

// ------------------------- conv3x3 implicit GEMM ---------------------------
// Block = 8 waves sharing one 32-wide N-group; wave w handles M-tile
// mtBase + w. Per K-chunk: async-stage both 16-wide B sub-tiles (9 taps,
// 18KB) into LDS, s_wait_asynccnt, barrier, then 9 taps x 2 WMMAs per chunk.
// A fragment (16x32 f16): lane m = lane&15, kbase=(lane>>4)*8,
//   elems 0..7 <-> K=kbase..+7 ; elems 8..15 <-> K=16+kbase..+7.
template <int CINP, int COUT>
__global__ __launch_bounds__(256) void conv3x3_wmma(
    const half_t* __restrict__ act, const half_t* __restrict__ wf,
    const float* __restrict__ bias, float* __restrict__ out,
    int Bn, int H) {
  static_assert(CINP % 32 == 0 && COUT % 32 == 0, "pad dims");
  constexpr int CHUNKS = CINP / 32;
  constexpr int NG = COUT / 32;
  __shared__ __attribute__((aligned(64))) half_t lds_b[2 * 9 * 512];  // 18KB

  const int W = H;
  const int NPIX = Bn * H * W;
  const int mtiles = (NPIX + 15) >> 4;
  const int ng = blockIdx.x % NG;
  const int mtBase = (blockIdx.x / NG) * 8;
  const int wave = threadIdx.x >> 5;     // wave32: 8 waves / 256 threads
  const int lane = threadIdx.x & 31;
  const int mcol = lane & 15;            // A: M row ; B/D: N col
  const int kbase = (lane >> 4) << 3;    // A: K sub-offset
  const int mt = mtBase + wave;

  int  p  = mt * 16 + mcol;
  bool pv = (mt < mtiles) && (p < NPIX);
  int  pc  = pv ? p : 0;
  int  bi  = pc / (H * W);
  int  rem = pc - bi * H * W;
  int  hh  = rem / W, ww = rem - hh * W;

  const int nt0 = ng * 2;
  v8f acc0 = {}, acc1 = {};
  const v8h z8 = {};

  for (int c = 0; c < CHUNKS; ++c) {
    __syncthreads();  // previous chunk fully consumed before overwrite
    {
      // Stage B fragments for both 16-wide sub-tiles: 2 x 9KB contiguous.
      const half_t* src0 = wf + ((size_t)(nt0 * CHUNKS + c) * 9) * 512;
      const half_t* src1 = wf + ((size_t)((nt0 + 1) * CHUNKS + c) * 9) * 512;
      for (int j = threadIdx.x; j < 1152; j += 256) {
        const half_t* g = (j < 576) ? (src0 + (size_t)j * 8)
                                    : (src1 + (size_t)(j - 576) * 8);
        unsigned ldsoff = (unsigned)(size_t)&lds_b[j * 8];  // LDS byte offset
        asm volatile("global_load_async_to_lds_b128 %0, %1, off"
                     :: "v"(ldsoff), "v"((unsigned long long)(size_t)g)
                     : "memory");
      }
      asm volatile("s_wait_asynccnt 0x0" ::: "memory");  // my loads resident
    }
    __syncthreads();  // all waves' stages resident

    const int cko = c * 32 + kbase;
    if (pv && c + 1 < CHUNKS)  // global_prefetch_b8 of next K-chunk
      __builtin_prefetch(act + ((size_t)(bi * H + hh) * W + ww) * CINP + cko + 32, 0, 3);
#pragma unroll
    for (int t = 0; t < 9; ++t) {
      const int  ih = hh + t / 3 - 1, iw = ww + t % 3 - 1;
      const bool ok = pv && (unsigned)ih < (unsigned)H && (unsigned)iw < (unsigned)W;
      union { v16h v; v8h h[2]; } au;
      if (ok) {
        const half_t* ap = act + ((size_t)(bi * H + ih) * W + iw) * CINP + cko;
        au.h[0] = *(const v8h*)ap;         // K = kbase .. kbase+7
        au.h[1] = *(const v8h*)(ap + 16);  // K = 16+kbase .. 16+kbase+7
      } else {
        au.h[0] = z8; au.h[1] = z8;        // halo / tail rows -> zeros
      }
      v16h b0 = *(const v16h*)&lds_b[t * 512 + lane * 16];         // ds_load
      v16h b1 = *(const v16h*)&lds_b[4608 + t * 512 + lane * 16];  // ds_load
      acc0 = __builtin_amdgcn_wmma_f32_16x16x32_f16(false, au.v, false, b0,
                                                    (short)0, acc0, false, false);
      acc1 = __builtin_amdgcn_wmma_f32_16x16x32_f16(false, au.v, false, b1,
                                                    (short)0, acc1, false, false);
    }
  }

  // Epilogue: bias fused; D layout lane n = lane&15, row m = (lane>>4)*8 + r.
  const float bb0 = bias[nt0 * 16 + mcol];
  const float bb1 = bias[nt0 * 16 + 16 + mcol];
  const int   m0  = (lane >> 4) << 3;
#pragma unroll
  for (int r = 0; r < 8; ++r) {
    int pm = mt * 16 + m0 + r;
    if (mt < mtiles && pm < NPIX) {
      out[(size_t)pm * COUT + nt0 * 16 + mcol]      = acc0[r] + bb0;
      out[(size_t)pm * COUT + nt0 * 16 + 16 + mcol] = acc1[r] + bb1;
    }
  }
}

// ------------------------- instance-norm + ReLU + f16 ----------------------
__global__ void k_inorm_relu(const float* __restrict__ src, half_t* __restrict__ dst,
                             int C, int Cd, int co, int HW) {
  __shared__ float ssum[256], ssq[256];
  int b = blockIdx.x / C, ch = blockIdx.x % C;
  const float* s = src + (size_t)b * HW * C + ch;
  float sum = 0.f, sq = 0.f;
  for (int i = threadIdx.x; i < HW; i += 256) {
    float v = s[(size_t)i * C];
    sum += v; sq += v * v;
  }
  ssum[threadIdx.x] = sum; ssq[threadIdx.x] = sq;
  __syncthreads();
  for (int off = 128; off > 0; off >>= 1) {
    if (threadIdx.x < off) {
      ssum[threadIdx.x] += ssum[threadIdx.x + off];
      ssq[threadIdx.x]  += ssq[threadIdx.x + off];
    }
    __syncthreads();
  }
  float mean = ssum[0] / HW;
  float var  = ssq[0] / HW - mean * mean;
  float rs   = rsqrtf(var + 1e-5f);
  half_t* d = dst + (size_t)b * HW * Cd + co + ch;
  for (int i = threadIdx.x; i < HW; i += 256) {
    float v = (s[(size_t)i * C] - mean) * rs;
    d[(size_t)i * Cd] = (half_t)fmaxf(v, 0.f);
  }
}

// ------------------------- 2x2 maxpool (f16 NHWC) --------------------------
__global__ void k_maxpool(const half_t* __restrict__ src, half_t* __restrict__ dst,
                          int Cs, int co, int C, int H, int n) {
  int t = blockIdx.x * blockDim.x + threadIdx.x;
  if (t >= n) return;
  int Ho = H >> 1;
  int c = t % C; int r = t / C;
  int wo = r % Ho; r /= Ho;
  int ho = r % Ho; int b = r / Ho;
  float m = -3.4e38f;
#pragma unroll
  for (int dy = 0; dy < 2; ++dy)
#pragma unroll
    for (int dx = 0; dx < 2; ++dx) {
      float v = (float)src[((size_t)(b * H + 2 * ho + dy) * H + 2 * wo + dx) * Cs + co + c];
      m = fmaxf(m, v);
    }
  dst[((size_t)(b * Ho + ho) * Ho + wo) * C + c] = (half_t)m;
}

// ------------------------- bilinear x2 upsample ----------------------------
__global__ void k_upsample(const half_t* __restrict__ src, half_t* __restrict__ dst,
                           int C, int H, int Cd, int n) {
  int t = blockIdx.x * blockDim.x + threadIdx.x;
  if (t >= n) return;
  int Ho = H * 2;
  int c = t % C; int r = t / C;
  int wo = r % Ho; r /= Ho;
  int ho = r % Ho; int b = r / Ho;
  float fy = fminf(fmaxf((ho + 0.5f) * 0.5f - 0.5f, 0.f), (float)(H - 1));
  float fx = fminf(fmaxf((wo + 0.5f) * 0.5f - 0.5f, 0.f), (float)(H - 1));
  int y0 = (int)fy, x0 = (int)fx;
  int y1 = min(y0 + 1, H - 1), x1 = min(x0 + 1, H - 1);
  float wy = fy - y0, wx = fx - x0;
  auto at = [&](int y, int x) {
    return (float)src[((size_t)(b * H + y) * H + x) * C + c];
  };
  float v = (1.f - wy) * ((1.f - wx) * at(y0, x0) + wx * at(y0, x1)) +
            wy        * ((1.f - wx) * at(y1, x0) + wx * at(y1, x1));
  dst[((size_t)(b * Ho + ho) * Ho + wo) * Cd + c] = (half_t)v;
}

// ------------------------- 1x1 conv head (32->16->1->1) --------------------
__global__ void k_head(const half_t* __restrict__ x,
                       const float* __restrict__ w1, const float* __restrict__ b1,
                       const float* __restrict__ w2, const float* __restrict__ b2,
                       const float* __restrict__ w3, const float* __restrict__ b3,
                       float* __restrict__ out, int n) {
  int t = blockIdx.x * blockDim.x + threadIdx.x;
  if (t >= n) return;
  const half_t* xp = x + (size_t)t * 32;
  float xs[32];
#pragma unroll
  for (int i = 0; i < 32; ++i) xs[i] = (float)xp[i];
  float y2 = b2[0];
  for (int o = 0; o < 16; ++o) {
    float s = b1[o];
#pragma unroll
    for (int i = 0; i < 32; ++i) s += w1[o * 32 + i] * xs[i];
    y2 += w2[o] * s;
  }
  out[t] = w3[0] * y2 + b3[0];
}

// ------------------------- tail: scatter + argmax mask ---------------------
__global__ void k_zero(float* __restrict__ p, int n) {
  int t = blockIdx.x * blockDim.x + threadIdx.x;
  if (t < n) p[t] = 0.f;
}

__global__ void k_scatter(const float* __restrict__ samples, const int* __restrict__ idx,
                          float* __restrict__ sel, int ns, int HW, int W) {
  int t = blockIdx.x * blockDim.x + threadIdx.x;
  if (t >= ns * 8) return;
  int b = t / ns, s = t % ns;
  int r = idx[2 * s], c = idx[2 * s + 1];
  size_t k = (size_t)b * HW + (size_t)r * W + c;
  sel[k] = samples[k];
}

__global__ void k_argmax_mask(const float* __restrict__ sel, float* __restrict__ out,
                              int HW) {
  __shared__ float sv[256];
  __shared__ int   si[256];
  int b = blockIdx.x;
  float best = -3.4e38f; int bi = 0x7fffffff;
  for (int i = threadIdx.x; i < HW; i += 256) {
    float v = sel[(size_t)b * HW + i];
    if (v > best) { best = v; bi = i; }   // i ascends -> keeps first max
  }
  sv[threadIdx.x] = best; si[threadIdx.x] = bi;
  __syncthreads();
  for (int off = 128; off > 0; off >>= 1) {
    if (threadIdx.x < off) {
      float v2 = sv[threadIdx.x + off]; int i2 = si[threadIdx.x + off];
      if (v2 > sv[threadIdx.x] || (v2 == sv[threadIdx.x] && i2 < si[threadIdx.x])) {
        sv[threadIdx.x] = v2; si[threadIdx.x] = i2;
      }
    }
    __syncthreads();
  }
  if (threadIdx.x == 0 && si[0] < HW) out[(size_t)b * HW + si[0]] = sv[0];
}

// ===========================================================================
extern "C" void kernel_launch(void* const* d_in, const int* in_sizes, int n_in,
                              void* d_out, int out_size, void* d_ws, size_t ws_size,
                              hipStream_t stream) {
  (void)in_sizes; (void)n_in; (void)ws_size;
  const int B = 8;
  const int HW0 = 320 * 320, HW1 = 160 * 160, HW2 = 80 * 80, HW3 = 40 * 40, HW4 = 20 * 20;

  // -------- input pointers (dict order; params leaves in sorted pytree order)
  const float* ks   = (const float*)d_in[0];
  const float* c2b1 = (const float*)d_in[1];
  const float* c2b2 = (const float*)d_in[2];
  const float* c2b3 = (const float*)d_in[3];
  const float* c2w1 = (const float*)d_in[4];
  const float* c2w2 = (const float*)d_in[5];
  const float* c2w3 = (const float*)d_in[6];
  const float* bptr[18] = {
    (const float*)d_in[7],  (const float*)d_in[8],
    (const float*)d_in[11], (const float*)d_in[12],
    (const float*)d_in[15], (const float*)d_in[16],
    (const float*)d_in[19], (const float*)d_in[20],
    (const float*)d_in[23], (const float*)d_in[24],
    (const float*)d_in[27], (const float*)d_in[28],
    (const float*)d_in[31], (const float*)d_in[32],
    (const float*)d_in[35], (const float*)d_in[36],
    (const float*)d_in[39], (const float*)d_in[40]};
  const float* wptr[18] = {
    (const float*)d_in[9],  (const float*)d_in[10],
    (const float*)d_in[13], (const float*)d_in[14],
    (const float*)d_in[17], (const float*)d_in[18],
    (const float*)d_in[21], (const float*)d_in[22],
    (const float*)d_in[25], (const float*)d_in[26],
    (const float*)d_in[29], (const float*)d_in[30],
    (const float*)d_in[33], (const float*)d_in[34],
    (const float*)d_in[37], (const float*)d_in[38],
    (const float*)d_in[41], (const float*)d_in[42]};
  const int* indexes = (const int*)d_in[43];

  struct CC { int cinp, cout, cin; };
  const CC cc[18] = {
    {32, 32, 2},    {32, 32, 32},     // down0
    {32, 64, 32},   {64, 64, 64},     // down1
    {64, 128, 64},  {128, 128, 128},  // down2
    {128, 256, 128},{256, 256, 256},  // down3
    {256, 256, 256},{256, 256, 256},  // mid
    {512, 128, 512},{128, 128, 128},  // up0
    {256, 64, 256}, {64, 64, 64},     // up1
    {128, 32, 128}, {32, 32, 32},     // up2
    {64, 32, 64},   {32, 32, 32}};    // up3

  // -------- workspace carve (bump allocator, 256B aligned)
  char* base = (char*)d_ws;
  size_t off = 0;
  auto alloc = [&](size_t bytes) {
    size_t r = off; off += (bytes + 255) & ~(size_t)255; return base + r;
  };
  half_t* x_in  = (half_t*)alloc((size_t)B * HW0 * 32 * 2);
  half_t* tmp16 = (half_t*)alloc((size_t)B * HW0 * 32 * 2);
  half_t* cat3  = (half_t*)alloc((size_t)B * HW0 * 64 * 2);
  half_t* cat2  = (half_t*)alloc((size_t)B * HW1 * 128 * 2);
  half_t* cat1  = (half_t*)alloc((size_t)B * HW2 * 256 * 2);
  half_t* cat0  = (half_t*)alloc((size_t)B * HW3 * 512 * 2);
  half_t* p0    = (half_t*)alloc((size_t)B * HW1 * 32 * 2);
  half_t* p1    = (half_t*)alloc((size_t)B * HW2 * 64 * 2);
  half_t* p2    = (half_t*)alloc((size_t)B * HW3 * 128 * 2);
  half_t* p3    = (half_t*)alloc((size_t)B * HW4 * 256 * 2);
  half_t* midb  = (half_t*)alloc((size_t)B * HW4 * 256 * 2);
  half_t* u0    = (half_t*)alloc((size_t)B * HW3 * 128 * 2);
  half_t* u1    = (half_t*)alloc((size_t)B * HW2 * 64 * 2);
  half_t* u2    = (half_t*)alloc((size_t)B * HW1 * 32 * 2);
  half_t* u3    = (half_t*)alloc((size_t)B * HW0 * 32 * 2);
  size_t wfoff[18], wtot = 0;
  for (int i = 0; i < 18; ++i) {
    wfoff[i] = wtot;
    wtot += (size_t)(cc[i].cout / 16) * 9 * (cc[i].cinp / 32) * 512;
  }
  half_t* wfrag    = (half_t*)alloc(wtot * 2);
  float*  convs    = (float*)alloc((size_t)B * HW0 * 32 * 4);
  float*  samples  = (float*)alloc((size_t)B * HW0 * 4);
  float*  selected = (float*)alloc((size_t)B * HW0 * 4);

  auto g1 = [](int n) { return (n + 255) / 256; };
  auto cgrid = [&](int HW, int cout) {
    int mtiles = (B * HW + 15) / 16;
    return (cout / 32) * ((mtiles + 7) / 8);
  };

  // -------- stage 0: input conversion + weight pre-pack
  k_convert_input<<<g1(B * HW0 * 32), 256, 0, stream>>>(ks, x_in, B * HW0);
  for (int i = 0; i < 18; ++i) {
    int tot = (cc[i].cout / 16) * 9 * (cc[i].cinp / 32) * 512;
    k_prep_w<<<g1(tot), 256, 0, stream>>>(wptr[i], wfrag + wfoff[i],
                                          cc[i].cout, cc[i].cin, cc[i].cinp / 32, tot);
  }

#define CONV(CIN, COUT, IN, I, OUT, HWv, Hv)                                      \
  conv3x3_wmma<CIN, COUT><<<cgrid(HWv, COUT), 256, 0, stream>>>(                  \
      (IN), wfrag + wfoff[I], bptr[I], (OUT), B, Hv)
#define INORM(SRC, DST, Cv, Cdv, cov, HWv)                                        \
  k_inorm_relu<<<B * (Cv), 256, 0, stream>>>((SRC), (DST), Cv, Cdv, cov, HWv)

  // -------- down path (block outputs land in the skip half of cat buffers)
  CONV(32, 32, x_in, 0, convs, HW0, 320);  INORM(convs, tmp16, 32, 32, 0, HW0);
  CONV(32, 32, tmp16, 1, convs, HW0, 320); INORM(convs, cat3, 32, 64, 32, HW0);
  k_maxpool<<<g1(B * 32 * HW1), 256, 0, stream>>>(cat3, p0, 64, 32, 32, 320, B * 32 * HW1);

  CONV(32, 64, p0, 2, convs, HW1, 160);    INORM(convs, tmp16, 64, 64, 0, HW1);
  CONV(64, 64, tmp16, 3, convs, HW1, 160); INORM(convs, cat2, 64, 128, 64, HW1);
  k_maxpool<<<g1(B * 64 * HW2), 256, 0, stream>>>(cat2, p1, 128, 64, 64, 160, B * 64 * HW2);

  CONV(64, 128, p1, 4, convs, HW2, 80);     INORM(convs, tmp16, 128, 128, 0, HW2);
  CONV(128, 128, tmp16, 5, convs, HW2, 80); INORM(convs, cat1, 128, 256, 128, HW2);
  k_maxpool<<<g1(B * 128 * HW3), 256, 0, stream>>>(cat1, p2, 256, 128, 128, 80, B * 128 * HW3);

  CONV(128, 256, p2, 6, convs, HW3, 40);    INORM(convs, tmp16, 256, 256, 0, HW3);
  CONV(256, 256, tmp16, 7, convs, HW3, 40); INORM(convs, cat0, 256, 512, 256, HW3);
  k_maxpool<<<g1(B * 256 * HW4), 256, 0, stream>>>(cat0, p3, 512, 256, 256, 40, B * 256 * HW4);

  // -------- mid
  CONV(256, 256, p3, 8, convs, HW4, 20);    INORM(convs, tmp16, 256, 256, 0, HW4);
  CONV(256, 256, tmp16, 9, convs, HW4, 20); INORM(convs, midb, 256, 256, 0, HW4);

  // -------- up path (upsample into low half of cat buffer, then conv block)
  k_upsample<<<g1(B * 256 * HW3), 256, 0, stream>>>(midb, cat0, 256, 20, 512, B * 256 * HW3);
  CONV(512, 128, cat0, 10, convs, HW3, 40);  INORM(convs, tmp16, 128, 128, 0, HW3);
  CONV(128, 128, tmp16, 11, convs, HW3, 40); INORM(convs, u0, 128, 128, 0, HW3);

  k_upsample<<<g1(B * 128 * HW2), 256, 0, stream>>>(u0, cat1, 128, 40, 256, B * 128 * HW2);
  CONV(256, 64, cat1, 12, convs, HW2, 80);  INORM(convs, tmp16, 64, 64, 0, HW2);
  CONV(64, 64, tmp16, 13, convs, HW2, 80);  INORM(convs, u1, 64, 64, 0, HW2);

  k_upsample<<<g1(B * 64 * HW1), 256, 0, stream>>>(u1, cat2, 64, 80, 128, B * 64 * HW1);
  CONV(128, 32, cat2, 14, convs, HW1, 160); INORM(convs, tmp16, 32, 32, 0, HW1);
  CONV(32, 32, tmp16, 15, convs, HW1, 160); INORM(convs, u2, 32, 32, 0, HW1);

  k_upsample<<<g1(B * 32 * HW0), 256, 0, stream>>>(u2, cat3, 32, 160, 64, B * 32 * HW0);
  CONV(64, 32, cat3, 16, convs, HW0, 320);  INORM(convs, tmp16, 32, 32, 0, HW0);
  CONV(32, 32, tmp16, 17, convs, HW0, 320); INORM(convs, u3, 32, 32, 0, HW0);

  // -------- head + sampling + argmax mask
  k_head<<<g1(B * HW0), 256, 0, stream>>>(u3, c2w1, c2b1, c2w2, c2b2, c2w3, c2b3,
                                          samples, B * HW0);
  k_zero<<<g1(B * HW0), 256, 0, stream>>>(selected, B * HW0);
  k_zero<<<g1(out_size), 256, 0, stream>>>((float*)d_out, out_size);
  k_scatter<<<g1(5000 * 8), 256, 0, stream>>>(samples, indexes, selected, 5000, HW0, 320);
  k_argmax_mask<<<B, 256, 0, stream>>>(selected, (float*)d_out, HW0);

#undef CONV
#undef INORM
}
